// MoEBlock_37503654429360
// MI455X (gfx1250) — matrix operand order repaired and checked by reading, not compiled, not run on hardware
//
#include <hip/hip_runtime.h>
#include <hip/hip_bf16.h>

typedef __attribute__((ext_vector_type(16))) __bf16 v16bf;
typedef __attribute__((ext_vector_type(8)))  __bf16 v8bf;
typedef __attribute__((ext_vector_type(8)))  float  v8f;
typedef __attribute__((ext_vector_type(4)))  float  f32x4;   // native vector: NT-builtin friendly

#define C_DIM 128
#define NUM_E 8
#define THW   32768            // 8*64*64
#define NVOX  65536            // B * THW
#define STRIDE_X 20            // padded fp32 row stride (floats) to dodge LDS bank conflicts

// per-wave LDS layout (bytes)
#define X32_BYTES  (C_DIM * STRIDE_X * 4)   // 10240: fp32 x tile [c][m]
#define XBF_BYTES  (16 * C_DIM * 2)         //  4096: bf16 x tile [m][c] (A-frag layout)
#define HBF_BYTES  (16 * C_DIM * 2)         //  4096: bf16 h tile [m][d]
#define WGT_BYTES  (16 * NUM_E * 4)         //   512: gate weights [m][e]
#define WAVE_LDS_BYTES (X32_BYTES + XBF_BYTES + HBF_BYTES + WGT_BYTES)
#define WAVES_PER_BLOCK 4
#define BLOCK_LDS_BYTES (WAVES_PER_BLOCK * WAVE_LDS_BYTES + C_DIM * NUM_E * 4)

// fast sigmoid: bf16-bound output, so v_rcp_f32 (1 ulp) is plenty
__device__ __forceinline__ float fast_sigmoid(float v) {
    return __builtin_amdgcn_rcpf(1.f + __expf(-v));
}

// ---- fragment loaders -------------------------------------------------------
// A (16x32 bf16): lanes 0-15 = rows M, elems 0..7 -> K = f*32+{0..7} (+8 for hi lanes),
// elems 8..15 -> +16.  buf layout: [m][c] contiguous in c.
__device__ __forceinline__ v16bf load_fragA(const __bf16* buf, int nn, int hi, int f) {
    const __bf16* p = buf + nn * C_DIM + f * 32 + hi * 8;
    v8bf lo = *(const v8bf*)p;
    v8bf h8 = *(const v8bf*)(p + 16);
    v16bf r;
#pragma unroll
    for (int i = 0; i < 8; ++i) { r[i] = lo[i]; r[i + 8] = h8[i]; }
    return r;
}

// B (32x16 bf16): lane = column N, same K order as A: elems 0..7 -> K = f*32+{0..7}
// (+8 for hi lanes), elems 8..15 -> +16. row = &W[n][0].
__device__ __forceinline__ v16bf load_fragB(const __bf16* row, int hi, int f) {
    const __bf16* p = row + f * 32 + hi * 8;
    v8bf lo = *(const v8bf*)p;
    v8bf h8 = *(const v8bf*)(p + 16);
    v16bf r;
#pragma unroll
    for (int i = 0; i < 8; ++i) { r[i] = lo[i]; r[i + 8] = h8[i]; }
    return r;
}

// ---- weight fp32 -> bf16 pre-conversion (runs once per launch, lands in L2) --
__global__ void cvt_weights_kernel(const float* __restrict__ w1,
                                   const float* __restrict__ w2,
                                   __bf16* __restrict__ ws) {
    int i = blockIdx.x * blockDim.x + threadIdx.x;   // 0 .. 131071
    if (i < C_DIM * C_DIM * NUM_E) {
        ws[i]                         = (__bf16)w1[i];
        ws[C_DIM * C_DIM * NUM_E + i] = (__bf16)w2[i];
    }
}

// ---- main fused MoE kernel --------------------------------------------------
__global__ void __launch_bounds__(128, 1) moe_kernel(
    const float* __restrict__ x,  const float* __restrict__ gate_w,
    const float* __restrict__ gb, const float* __restrict__ b1,
    const float* __restrict__ b2, const __bf16* __restrict__ w1bf,
    const __bf16* __restrict__ w2bf, float* __restrict__ out) {
    extern __shared__ char smem[];
    const int tid  = threadIdx.x;
    const int lane = tid & 31;
    const int wid  = tid >> 5;
    const int n    = lane & 15;   // column within 16-wide tiles
    const int hi   = lane >> 4;   // half-wave select

    char*   wbase = smem + wid * WAVE_LDS_BYTES;
    float*  x32   = (float*)wbase;
    __bf16* xbf   = (__bf16*)(wbase + X32_BYTES);
    __bf16* hbf   = (__bf16*)(wbase + X32_BYTES + XBF_BYTES);
    float*  wgt   = (float*)(wbase + X32_BYTES + XBF_BYTES + HBF_BYTES);
    float*  ldsgw = (float*)(smem + WAVES_PER_BLOCK * WAVE_LDS_BYTES); // gate_w^T [c][e]

    // stage gate_w transposed (block-wide, 4 KB)
    for (int i = tid; i < C_DIM * NUM_E; i += blockDim.x) {
        int c = i >> 3, e = i & 7;
        ldsgw[i] = gate_w[e * C_DIM + c];
    }
    __syncthreads();

    // this wave's 16-voxel tile
    const int gwave = blockIdx.x * WAVES_PER_BLOCK + wid;
    const int v0 = gwave * 16;
    const int bb = v0 / THW;
    const int s0 = v0 % THW;
    const float* xg = x   + (size_t)bb * C_DIM * THW + s0;
    float*       og = out + (size_t)bb * C_DIM * THW + s0;

    // ---- coalesced x load (non-temporal: streamed once): fp32 -> x32[c][m],
    //      bf16 -> xbf[m][c] ----
    {
        const int cl = lane >> 2;
        const int v4 = (lane & 3) * 4;
#pragma unroll
        for (int it = 0; it < 16; ++it) {
            int c = it * 8 + cl;
            f32x4 v = __builtin_nontemporal_load(
                (const f32x4*)(xg + (size_t)c * THW + v4));
            *(f32x4*)(x32 + c * STRIDE_X + v4) = v;
            xbf[(v4 + 0) * C_DIM + c] = (__bf16)v.x;
            xbf[(v4 + 1) * C_DIM + c] = (__bf16)v.y;
            xbf[(v4 + 2) * C_DIM + c] = (__bf16)v.z;
            xbf[(v4 + 3) * C_DIM + c] = (__bf16)v.w;
        }
    }

    // ---- gate: fp32 logits, top-2 softmax, weights -> LDS ----
    if (lane < 16) {
        const int m = lane;
        float lg[NUM_E];
#pragma unroll
        for (int e = 0; e < NUM_E; ++e) lg[e] = gb[e];
        for (int c = 0; c < C_DIM; ++c) {
            float xv = x32[c * STRIDE_X + m];
            const float* g = ldsgw + c * NUM_E;
#pragma unroll
            for (int e = 0; e < NUM_E; ++e) lg[e] += xv * g[e];
        }
        int i1 = 0;
#pragma unroll
        for (int e = 1; e < NUM_E; ++e) if (lg[e] > lg[i1]) i1 = e;
        int i2 = (i1 == 0) ? 1 : 0;
#pragma unroll
        for (int e = 0; e < NUM_E; ++e) if (e != i1 && lg[e] > lg[i2]) i2 = e;
        float p1 = fast_sigmoid(lg[i1] - lg[i2]);   // softmax over top-2
#pragma unroll
        for (int e = 0; e < NUM_E; ++e) wgt[m * NUM_E + e] = 0.f;
        wgt[m * NUM_E + i1] = p1;
        wgt[m * NUM_E + i2] = 1.f - p1;
    }
    // (divergence closed; EXEC is all-ones again for the WMMA region)

    // x A-fragments, reused by all 8 experts
    v16bf xa[4];
#pragma unroll
    for (int f = 0; f < 4; ++f) xa[f] = load_fragA(xbf, n, hi, f);

    v8f oacc[8];
#pragma unroll
    for (int ct = 0; ct < 8; ++ct)
#pragma unroll
        for (int r = 0; r < 8; ++r) oacc[ct][r] = 0.f;

    for (int e = 0; e < NUM_E; ++e) {
        // -------- layer 1: h_e = silu(x @ w1_e^T + b1_e), 8x4 WMMAs --------
        // unroll 2: two independent WMMA accumulator chains interleave, hiding
        // XDL RAW latency and B-fragment load latency within one wave
#pragma unroll 2
        for (int j = 0; j < 8; ++j) {
            int o = e * C_DIM + j * 16 + n;
            float bv = b1[o];
            v8f acc;
#pragma unroll
            for (int r = 0; r < 8; ++r) acc[r] = bv;
            const __bf16* wrow = w1bf + (size_t)o * C_DIM;
#pragma unroll
            for (int f = 0; f < 4; ++f) {
                v16bf bf = load_fragB(wrow, hi, f);
                acc = __builtin_amdgcn_wmma_f32_16x16x32_bf16(
                    false, xa[f], false, bf, (short)0, acc, false, false);
            }
#pragma unroll
            for (int r = 0; r < 8; ++r) {
                float v = acc[r];
                float s = v * fast_sigmoid(v);       // SiLU, rcp-based
                hbf[(r + hi * 8) * C_DIM + (j * 16 + n)] = (__bf16)s;
            }
        }
        // re-layout h_e D-tiles into A-fragments via LDS
        v16bf ha[4];
#pragma unroll
        for (int f = 0; f < 4; ++f) ha[f] = load_fragA(hbf, n, hi, f);
        // per-row gate weights for this expert
        float wv[8];
#pragma unroll
        for (int r = 0; r < 8; ++r) wv[r] = wgt[(r + hi * 8) * NUM_E + e];

        // -------- layer 2: eo_e = h_e @ w2_e^T + b2_e, weighted-accumulate --------
#pragma unroll 2
        for (int ct = 0; ct < 8; ++ct) {
            int co = ct * 16 + n;
            float bv = b2[e * C_DIM + co];
            v8f acc;
#pragma unroll
            for (int r = 0; r < 8; ++r) acc[r] = bv;
            const __bf16* wrow = w2bf + ((size_t)e * C_DIM + co) * C_DIM;
#pragma unroll
            for (int f = 0; f < 4; ++f) {
                v16bf bf = load_fragB(wrow, hi, f);
                acc = __builtin_amdgcn_wmma_f32_16x16x32_bf16(
                    false, ha[f], false, bf, (short)0, acc, false, false);
            }
#pragma unroll
            for (int r = 0; r < 8; ++r) oacc[ct][r] += wv[r] * acc[r];
        }
    }

    // ---- residual add into fp32 LDS tile, then coalesced non-temporal store ----
#pragma unroll
    for (int ct = 0; ct < 8; ++ct) {
#pragma unroll
        for (int r = 0; r < 8; ++r) {
            int c = ct * 16 + n;
            int m = r + hi * 8;
            x32[c * STRIDE_X + m] += oacc[ct][r];
        }
    }
    {
        const int cl = lane >> 2;
        const int v4 = (lane & 3) * 4;
#pragma unroll
        for (int it = 0; it < 16; ++it) {
            int c = it * 8 + cl;
            f32x4 v = *(const f32x4*)(x32 + c * STRIDE_X + v4);
            __builtin_nontemporal_store(v, (f32x4*)(og + (size_t)c * THW + v4));
        }
    }
}

extern "C" void kernel_launch(void* const* d_in, const int* in_sizes, int n_in,
                              void* d_out, int out_size, void* d_ws, size_t ws_size,
                              hipStream_t stream) {
    const float* x      = (const float*)d_in[0];
    const float* gate_w = (const float*)d_in[1];
    const float* gate_b = (const float*)d_in[2];
    const float* w1     = (const float*)d_in[3];
    const float* b1     = (const float*)d_in[4];
    const float* w2     = (const float*)d_in[5];
    const float* b2     = (const float*)d_in[6];
    float* out = (float*)d_out;

    __bf16* ws   = (__bf16*)d_ws;
    __bf16* w1bf = ws;
    __bf16* w2bf = ws + C_DIM * C_DIM * NUM_E;

    // 1) bf16 weight staging (512 KB, stays L2-resident)
    cvt_weights_kernel<<<(C_DIM * C_DIM * NUM_E + 255) / 256, 256, 0, stream>>>(w1, w2, ws);

    // 2) fused MoE: 1 wave per 16 voxels, 4 waves/block
    const int nblocks = NVOX / (16 * WAVES_PER_BLOCK);   // 1024
    moe_kernel<<<nblocks, 32 * WAVES_PER_BLOCK, BLOCK_LDS_BYTES, stream>>>(
        x, gate_w, gate_b, b1, b2, w1bf, w2bf, out);
}
